// TransformerBlock_67714454388743
// MI455X (gfx1250) — matrix operand order repaired and checked by reading, not compiled, not run on hardware
//
#include <hip/hip_runtime.h>
#include <hip/hip_bf16.h>

// ---------------- problem constants ----------------
constexpr int kH   = 768;
constexpr int kMLP = 3072;
constexpr int kNH  = 12;
constexpr int kHD  = 64;
constexpr int kB   = 8;
constexpr int kS   = 1024;
constexpr int kTOK = kB * kS;

typedef __attribute__((ext_vector_type(16))) _Float16 v16h;
typedef __attribute__((ext_vector_type(8)))  _Float16 v8h;
typedef __attribute__((ext_vector_type(8)))  float    v8f;

// ---------------- WMMA helpers ----------------
static __device__ __forceinline__ v8f wmma_f16(v16h a, v16h b, v8f c) {
  // emits v_wmma_f32_16x16x32_f16
  return __builtin_amdgcn_wmma_f32_16x16x32_f16(false, a, false, b, (short)0, c,
                                                false, false);
}

// A fragment (16x32, MxK): lane L holds row m0+(L&15); K split pattern
//   elems 0..7  -> K = k0 + (L>>4)*8 + j
//   elems 8..15 -> K = k0 + 16 + (L>>4)*8 + j
static __device__ __forceinline__ v16h load_a(const _Float16* __restrict__ base,
                                              int ld, int m0, int k0, int lane) {
  int r  = m0 + (lane & 15);
  int kh = (lane >> 4) & 1;
  const _Float16* p = base + (size_t)r * ld + k0 + kh * 8;
  v8h lo = *(const v8h*)(p);
  v8h hi = *(const v8h*)(p + 16);
  v16h out;
#pragma unroll
  for (int j = 0; j < 8; ++j) { out[j] = lo[j]; out[j + 8] = hi[j]; }
  return out;
}

// B fragment (32x16, KxN) from row-major W[N,K]: lane L holds column n0+(L&15),
// elems 0..7 -> K = k0 + (L>>4)*8 + j, elems 8..15 -> K = k0 + 16 + (L>>4)*8 + j
static __device__ __forceinline__ v16h load_b(const _Float16* __restrict__ base,
                                              int ld, int n0, int k0, int lane) {
  int c  = n0 + (lane & 15);
  int kh = (lane >> 4) & 1;
  const _Float16* p = base + (size_t)c * ld + k0 + kh * 8;
  v8h lo = *(const v8h*)(p);
  v8h hi = *(const v8h*)(p + 16);
  v16h out;
#pragma unroll
  for (int j = 0; j < 8; ++j) { out[j] = lo[j]; out[j + 8] = hi[j]; }
  return out;
}

// ---------------- fp32 -> f16 convert ----------------
__global__ void cvt_f16_kernel(const float* __restrict__ src,
                               _Float16* __restrict__ dst, int n) {
  int i      = blockIdx.x * blockDim.x + threadIdx.x;
  int stride = gridDim.x * blockDim.x;
  for (; i < n; i += stride) dst[i] = (_Float16)src[i];
}

// ---------------- LayerNorm (wave per token) -> f16 ----------------
__global__ void __launch_bounds__(256) ln_f16_kernel(const float* __restrict__ x,
                                                     const float* __restrict__ g,
                                                     const float* __restrict__ bt,
                                                     _Float16* __restrict__ out) {
  int lane = threadIdx.x & 31;
  int wave = threadIdx.x >> 5;
  int tok  = blockIdx.x * 8 + wave;
  const float* xr = x + (size_t)tok * kH;
  float vals[24];
  float s = 0.f, s2 = 0.f;
#pragma unroll
  for (int j = 0; j < 24; ++j) {
    float v = xr[lane + 32 * j];
    vals[j] = v; s += v; s2 += v * v;
  }
#pragma unroll
  for (int m = 1; m < 32; m <<= 1) {
    s  += __shfl_xor(s,  m, 32);
    s2 += __shfl_xor(s2, m, 32);
  }
  float mu  = s * (1.0f / kH);
  float var = s2 * (1.0f / kH) - mu * mu;
  float inv = rsqrtf(var + 1e-5f);
  _Float16* orow = out + (size_t)tok * kH;
#pragma unroll
  for (int j = 0; j < 24; ++j) {
    int c = lane + 32 * j;
    orow[c] = (_Float16)((vals[j] - mu) * inv * g[c] + bt[c]);
  }
}

// ---------------- generic WMMA GEMM: C[M,N] = A[M,K] * W[N,K]^T ----------------
// mode 0: QKV scatter (q scaled, v transposed), mode 1: +bias+residual -> f32,
// mode 2: +bias, exact GELU -> f16
__global__ void __launch_bounds__(256)
gemm_wmma_kernel(const _Float16* __restrict__ A, const _Float16* __restrict__ W,
                 int M, int N, int K, int mode,
                 const float* __restrict__ bias, const float* __restrict__ res,
                 float* __restrict__ outf, _Float16* __restrict__ outh,
                 _Float16* __restrict__ qo, _Float16* __restrict__ ko,
                 _Float16* __restrict__ vto, float qscale) {
  int lane = threadIdx.x & 31;
  int wave = threadIdx.x >> 5;
  int wr = wave >> 1, wc = wave & 1;
  int m0 = blockIdx.y * 128 + wr * 32;
  int n0 = blockIdx.x * 128 + wc * 64;

  v8f z = {};
  v8f acc[2][4];
#pragma unroll
  for (int mi = 0; mi < 2; ++mi)
#pragma unroll
    for (int j = 0; j < 4; ++j) acc[mi][j] = z;

  for (int k0 = 0; k0 < K; k0 += 32) {
    v16h a0 = load_a(A, K, m0,      k0, lane);
    v16h a1 = load_a(A, K, m0 + 16, k0, lane);
#pragma unroll
    for (int j = 0; j < 4; ++j) {
      v16h b = load_b(W, K, n0 + j * 16, k0, lane);
      acc[0][j] = wmma_f16(a0, b, acc[0][j]);
      acc[1][j] = wmma_f16(a1, b, acc[1][j]);
    }
  }

  int cb = lane & 15;
  int rb = (lane >> 4) * 8;
#pragma unroll
  for (int mi = 0; mi < 2; ++mi) {
#pragma unroll
    for (int j = 0; j < 4; ++j) {
      int col = n0 + j * 16 + cb;
      if (mode == 0) {
        int qkv_i = col / kH;
        int wcol  = col - qkv_i * kH;
        int head  = wcol >> 6;
        int d     = wcol & 63;
#pragma unroll
        for (int i = 0; i < 8; ++i) {
          int row = m0 + mi * 16 + rb + i;
          int b   = row >> 10;
          int s   = row & 1023;
          int bh  = b * kNH + head;
          float v = acc[mi][j][i];
          if (qkv_i == 0)
            qo[((size_t)bh * kS + s) * kHD + d] = (_Float16)(v * qscale);
          else if (qkv_i == 1)
            ko[((size_t)bh * kS + s) * kHD + d] = (_Float16)v;
          else
            vto[((size_t)bh * kHD + d) * kS + s] = (_Float16)v;
        }
      } else if (mode == 1) {
        float bb = bias[col];
#pragma unroll
        for (int i = 0; i < 8; ++i) {
          int row = m0 + mi * 16 + rb + i;
          size_t idx = (size_t)row * N + col;
          outf[idx] = acc[mi][j][i] + bb + res[idx];
        }
      } else {
        float bb = bias[col];
#pragma unroll
        for (int i = 0; i < 8; ++i) {
          int row = m0 + mi * 16 + rb + i;
          float t = acc[mi][j][i] + bb;
          float gl = 0.5f * t * (1.0f + erff(t * 0.70710678118f));
          outh[(size_t)row * N + col] = (_Float16)gl;
        }
      }
    }
  }
}

// ---------------- flash attention (wave = 16 query rows) ----------------
__global__ void __launch_bounds__(128)
attn_kernel(const _Float16* __restrict__ q, const _Float16* __restrict__ k,
            const _Float16* __restrict__ vt, _Float16* __restrict__ o) {
  __shared__ _Float16 plds[4][16 * 32];
  int lane = threadIdx.x & 31;
  int wave = threadIdx.x >> 5;
  int bh   = blockIdx.y;
  int m0   = blockIdx.x * 64 + wave * 16;

  const _Float16* qp = q  + (size_t)bh * kS * kHD;
  const _Float16* kp = k  + (size_t)bh * kS * kHD;
  const _Float16* vp = vt + (size_t)bh * kHD * kS;

  v16h qa0 = load_a(qp, kHD, m0, 0,  lane);
  v16h qa1 = load_a(qp, kHD, m0, 32, lane);

  v8f z = {};
  v8f oacc[4];
#pragma unroll
  for (int t = 0; t < 4; ++t) oacc[t] = z;
  float mrow[8], lrow[8];
#pragma unroll
  for (int i = 0; i < 8; ++i) { mrow[i] = -1e30f; lrow[i] = 0.f; }

  _Float16* pl = plds[wave];
  int cb = lane & 15;
  int rb = (lane >> 4) * 8;
  int kh = (lane >> 4) * 8;

  for (int kb = 0; kb < kS; kb += 32) {
    // ---- scores: 16 queries x 32 keys, K-dim = 64 (head dim) ----
    v8f s0 = z, s1 = z;
    s0 = wmma_f16(qa0, load_b(kp, kHD, kb,      0,  lane), s0);
    s0 = wmma_f16(qa1, load_b(kp, kHD, kb,      32, lane), s0);
    s1 = wmma_f16(qa0, load_b(kp, kHD, kb + 16, 0,  lane), s1);
    s1 = wmma_f16(qa1, load_b(kp, kHD, kb + 16, 32, lane), s1);

    // ---- online softmax update ----
    float alpha[8], p0[8], p1[8];
#pragma unroll
    for (int i = 0; i < 8; ++i) {
      float cm = fmaxf(s0[i], s1[i]);
#pragma unroll
      for (int m = 1; m < 16; m <<= 1) cm = fmaxf(cm, __shfl_xor(cm, m, 32));
      float mn = fmaxf(mrow[i], cm);
      alpha[i] = __expf(mrow[i] - mn);
      p0[i]    = __expf(s0[i] - mn);
      p1[i]    = __expf(s1[i] - mn);
      float ps = p0[i] + p1[i];
#pragma unroll
      for (int m = 1; m < 16; m <<= 1) ps += __shfl_xor(ps, m, 32);
      lrow[i] = lrow[i] * alpha[i] + ps;
      mrow[i] = mn;
    }
#pragma unroll
    for (int t = 0; t < 4; ++t)
#pragma unroll
      for (int i = 0; i < 8; ++i) oacc[t][i] *= alpha[i];

    // ---- P: C-layout -> A-layout via per-wave LDS tile ----
#pragma unroll
    for (int i = 0; i < 8; ++i) {
      pl[(rb + i) * 32 + cb]      = (_Float16)p0[i];
      pl[(rb + i) * 32 + cb + 16] = (_Float16)p1[i];
    }
    asm volatile("s_wait_dscnt 0" ::: "memory");
    v16h pa;
#pragma unroll
    for (int j = 0; j < 8; ++j) {
      pa[j]     = pl[cb * 32 + kh + j];
      pa[j + 8] = pl[cb * 32 + 16 + kh + j];
    }
    asm volatile("s_wait_dscnt 0" ::: "memory");

    // ---- O += P * V  (V stored transposed [HD, S] -> contiguous B frags) ----
#pragma unroll
    for (int t = 0; t < 4; ++t)
      oacc[t] = wmma_f16(pa, load_b(vp, kS, t * 16, kb, lane), oacc[t]);
  }

  // ---- normalize + write O in [B,S,NH,HD] == [B*S, H] f16 layout ----
  int b    = bh / kNH;
  int head = bh % kNH;
#pragma unroll
  for (int i = 0; i < 8; ++i) {
    float inv = 1.0f / lrow[i];
    int s = m0 + rb + i;
    size_t base = ((size_t)b * kS + s) * kH + head * kHD;
#pragma unroll
    for (int t = 0; t < 4; ++t)
      o[base + t * 16 + cb] = (_Float16)(oacc[t][i] * inv);
  }
}

// ---------------- host side ----------------
extern "C" void kernel_launch(void* const* d_in, const int* in_sizes, int n_in,
                              void* d_out, int out_size, void* d_ws, size_t ws_size,
                              hipStream_t stream) {
  (void)in_sizes; (void)n_in; (void)out_size; (void)ws_size;
  const float* x    = (const float*)d_in[0];
  const float* ln1g = (const float*)d_in[1];
  const float* ln1b = (const float*)d_in[2];
  const float* wqkv = (const float*)d_in[3];
  const float* wout = (const float*)d_in[4];
  const float* bout = (const float*)d_in[5];
  const float* ln2g = (const float*)d_in[6];
  const float* ln2b = (const float*)d_in[7];
  const float* w1   = (const float*)d_in[8];
  const float* b1   = (const float*)d_in[9];
  const float* w2   = (const float*)d_in[10];
  const float* b2   = (const float*)d_in[11];
  float* out = (float*)d_out;

  char* ws = (char*)d_ws;
  size_t off = 0;
  auto alloc = [&](size_t bytes) -> char* {
    char* p = ws + off;
    off += (bytes + 255) & ~(size_t)255;
    return p;
  };
  _Float16* wqkv16 = (_Float16*)alloc((size_t)3 * kH * kH * 2);
  _Float16* wout16 = (_Float16*)alloc((size_t)kH * kH * 2);
  _Float16* w116   = (_Float16*)alloc((size_t)kMLP * kH * 2);
  _Float16* w216   = (_Float16*)alloc((size_t)kH * kMLP * 2);
  _Float16* hln    = (_Float16*)alloc((size_t)kTOK * kH * 2);  // -> reused as attn O
  _Float16* q16    = (_Float16*)alloc((size_t)kTOK * kH * 2);  // -> reused (w/ k16) as x1 f32
  _Float16* k16    = (_Float16*)alloc((size_t)kTOK * kH * 2);
  _Float16* vt16   = (_Float16*)alloc((size_t)kTOK * kH * 2);  // -> reused as hln2
  _Float16* hmlp   = (_Float16*)alloc((size_t)kTOK * kMLP * 2);
  float*    x1   = (float*)q16;     // 25.2 MB fits in q16+k16 region
  _Float16* hln2 = vt16;
  _Float16* oatt = hln;
  (void)k16;

  // 1) weights fp32 -> f16
  cvt_f16_kernel<<<512, 256, 0, stream>>>(wqkv, wqkv16, 3 * kH * kH);
  cvt_f16_kernel<<<256, 256, 0, stream>>>(wout, wout16, kH * kH);
  cvt_f16_kernel<<<512, 256, 0, stream>>>(w1, w116, kMLP * kH);
  cvt_f16_kernel<<<512, 256, 0, stream>>>(w2, w216, kH * kMLP);

  // 2) LN1 -> f16
  ln_f16_kernel<<<kTOK / 8, 256, 0, stream>>>(x, ln1g, ln1b, hln);

  // 3) QKV projection (scatter epilogue, q pre-scaled by HD^-0.5 = 0.125)
  dim3 gq(3 * kH / 128, kTOK / 128);
  gemm_wmma_kernel<<<gq, 256, 0, stream>>>(hln, wqkv16, kTOK, 3 * kH, kH, 0,
                                           nullptr, nullptr, nullptr, nullptr,
                                           q16, k16, vt16, 0.125f);

  // 4) flash attention
  dim3 ga(kS / 64, kB * kNH);
  attn_kernel<<<ga, 128, 0, stream>>>(q16, k16, vt16, oatt);

  // 5) output projection + b_out + residual(x) -> x1 (fp32)
  dim3 go(kH / 128, kTOK / 128);
  gemm_wmma_kernel<<<go, 256, 0, stream>>>(oatt, wout16, kTOK, kH, kH, 1,
                                           bout, x, x1, nullptr,
                                           nullptr, nullptr, nullptr, 0.f);

  // 6) LN2 -> f16
  ln_f16_kernel<<<kTOK / 8, 256, 0, stream>>>(x1, ln2g, ln2b, hln2);

  // 7) MLP up-proj + b1 + exact GELU -> f16
  dim3 gm(kMLP / 128, kTOK / 128);
  gemm_wmma_kernel<<<gm, 256, 0, stream>>>(hln2, w116, kTOK, kMLP, kH, 2,
                                           b1, nullptr, nullptr, hmlp,
                                           nullptr, nullptr, nullptr, 0.f);

  // 8) MLP down-proj + b2 + residual(x1) -> out (fp32)
  gemm_wmma_kernel<<<go, 256, 0, stream>>>(hmlp, w216, kTOK, kH, kMLP, 1,
                                           b2, x1, out, nullptr,
                                           nullptr, nullptr, nullptr, 0.f);
}
